// EdgeClassifier_13915694039646
// MI455X (gfx1250) — compile-verified, matrix operand order checked by reading, not run on hardware
//
#include <hip/hip_runtime.h>

// ---------------------------------------------------------------------------
// GatedGCN (3 layers) + edge MLP classifier for MI455X (gfx1250, wave32).
// All H x H and 384 x 128 GEMMs run on the matrix engine via
// v_wmma_f32_16x16x32_f16 (f16 inputs, f32 accumulate). Everything else
// (activations, BN, sigmoid gating, segment-sum via fp32 atomics) is fp32.
// ---------------------------------------------------------------------------

#define NNODES 20000
#define NEDGES 320000
#define HID    128

static __device__ __constant__ float kBN_S = 0.9999950000374997f; // 1/sqrt(1+1e-5)

typedef __attribute__((ext_vector_type(16))) _Float16 v16h;
typedef __attribute__((ext_vector_type(8)))  float    v8f;

__device__ __forceinline__ float lrelu(float x) { return x > 0.f ? x : 0.01f * x; }
__device__ __forceinline__ float elu_f(float x) { return x > 0.f ? x : (__expf(x) - 1.f); }
__device__ __forceinline__ float sigm(float x)  { return 1.f / (1.f + __expf(-x)); }

// ---------------------------------------------------------------------------
// WMMA tile loaders.
// A (16x32 f16, M x K): lane l holds row M = l&15; k-group kg = l>>4:
//   halfs 0..7  -> K = kg*8 .. kg*8+7
//   halfs 8..15 -> K = 16+kg*8 .. 16+kg*8+7
// B (32x16 f16, K x N): lane l holds column N = l&15 with the same K pattern,
// sourced from a pre-transposed f16 weight Wt[n][k] (contiguous in k).
// ---------------------------------------------------------------------------
__device__ __forceinline__ v16h load_a_f32(const float* __restrict__ Xrow, int kbase, int kg) {
  const float4* p0 = (const float4*)(Xrow + kbase + kg * 8);
  const float4* p1 = (const float4*)(Xrow + kbase + 16 + kg * 8);
  float4 f0 = p0[0], f1 = p0[1], f2 = p1[0], f3 = p1[1];
  v16h a;
  a[0]  = (_Float16)f0.x; a[1]  = (_Float16)f0.y; a[2]  = (_Float16)f0.z; a[3]  = (_Float16)f0.w;
  a[4]  = (_Float16)f1.x; a[5]  = (_Float16)f1.y; a[6]  = (_Float16)f1.z; a[7]  = (_Float16)f1.w;
  a[8]  = (_Float16)f2.x; a[9]  = (_Float16)f2.y; a[10] = (_Float16)f2.z; a[11] = (_Float16)f2.w;
  a[12] = (_Float16)f3.x; a[13] = (_Float16)f3.y; a[14] = (_Float16)f3.z; a[15] = (_Float16)f3.w;
  return a;
}

__device__ __forceinline__ v16h load_b_f16(const _Float16* __restrict__ Wtrow, int kbase, int kg) {
  union { uint4 u[2]; v16h h; } t;
  t.u[0] = *(const uint4*)(Wtrow + kbase + kg * 8);
  t.u[1] = *(const uint4*)(Wtrow + kbase + 16 + kg * 8);
  return t.h;
}

// ---------------------------------------------------------------------------
// Generic GEMM: Y[M,128] = X[M,128] @ W(128x128, as f16 Wt[n][k]) + bias.
// One wave -> one 16-row strip, all 8 column tiles (32 WMMAs / wave).
// ---------------------------------------------------------------------------
__global__ __launch_bounds__(256)
void gemm128_wmma(const float* __restrict__ X, const _Float16* __restrict__ Wt,
                  const float* __restrict__ bias, float* __restrict__ Y, int M) {
  const int wave = threadIdx.x >> 5;
  const int lane = threadIdx.x & 31;
  const int rowtile = blockIdx.x * 8 + wave;
  if (rowtile >= (M >> 4)) return;
  const int rl = lane & 15;
  const int kg = lane >> 4;
  const float* Xrow = X + (size_t)(rowtile * 16 + rl) * HID;

  v8f acc[8] = {};
#pragma unroll
  for (int ks = 0; ks < 4; ++ks) {
    const int kbase = ks * 32;
    v16h a = load_a_f32(Xrow, kbase, kg);
#pragma unroll
    for (int ct = 0; ct < 8; ++ct) {
      v16h b = load_b_f16(Wt + (size_t)(ct * 16 + rl) * HID, kbase, kg);
      acc[ct] = __builtin_amdgcn_wmma_f32_16x16x32_f16(
          false, a, false, b, (short)0, acc[ct], false, false);
    }
  }
  // C/D layout: lane holds N = rl; VGPR i holds M = i + 8*kg.
  const int mbase = rowtile * 16 + kg * 8;
#pragma unroll
  for (int ct = 0; ct < 8; ++ct) {
    const int n = ct * 16 + rl;
    const float bv = bias[n];
#pragma unroll
    for (int i = 0; i < 8; ++i)
      Y[(size_t)(mbase + i) * HID + n] = acc[ct][i] + bv;
  }
}

// ---------------------------------------------------------------------------
// Edge MLP layer 1: hid[E,128] = relu(concat(h[src],h[dst],e)[E,384] @ W1 + b1)
// Rows gathered on the fly; each 32-wide k-chunk stays inside one region.
// ---------------------------------------------------------------------------
__global__ __launch_bounds__(256)
void gemm_mlp_wmma(const float* __restrict__ h, const float* __restrict__ e,
                   const int* __restrict__ src, const int* __restrict__ dst,
                   const _Float16* __restrict__ Wt /*[128][384]*/,
                   const float* __restrict__ b1, float* __restrict__ hid, int E) {
  const int wave = threadIdx.x >> 5;
  const int lane = threadIdx.x & 31;
  const int rowtile = blockIdx.x * 8 + wave;
  if (rowtile >= (E >> 4)) return;
  const int rl = lane & 15;
  const int kg = lane >> 4;
  const int r = rowtile * 16 + rl;  // edge id for this lane
  const float* baseS = h + (size_t)src[r] * HID;
  const float* baseD = h + (size_t)dst[r] * HID;
  const float* baseE = e + (size_t)r * HID;

  v8f acc[8] = {};
#pragma unroll
  for (int ks = 0; ks < 12; ++ks) {
    const int kbase = ks * 32;
    const int region = kbase >> 7;       // 0: h[src], 1: h[dst], 2: e
    const int koff = kbase & 127;
    const float* Xrow = (region == 0) ? baseS : (region == 1) ? baseD : baseE;
    v16h a = load_a_f32(Xrow, koff, kg);
#pragma unroll
    for (int ct = 0; ct < 8; ++ct) {
      v16h b = load_b_f16(Wt + (size_t)(ct * 16 + rl) * 384, kbase, kg);
      acc[ct] = __builtin_amdgcn_wmma_f32_16x16x32_f16(
          false, a, false, b, (short)0, acc[ct], false, false);
    }
  }
  const int mbase = rowtile * 16 + kg * 8;
#pragma unroll
  for (int ct = 0; ct < 8; ++ct) {
    const int n = ct * 16 + rl;
    const float bv = b1[n];
#pragma unroll
    for (int i = 0; i < 8; ++i) {
      float v = acc[ct][i] + bv;
      hid[(size_t)(mbase + i) * HID + n] = v > 0.f ? v : 0.f;
    }
  }
}

// ---------------------------------------------------------------------------
// Small helper kernels
// ---------------------------------------------------------------------------
__global__ void cvt_wt_kernel(const float* __restrict__ W, _Float16* __restrict__ Wt,
                              int K, int Nc) {
  int gid = blockIdx.x * 256 + threadIdx.x;
  if (gid >= K * Nc) return;
  int k = gid / Nc, n = gid - k * Nc;
  Wt[(size_t)n * K + k] = (_Float16)W[gid];
}

__global__ void zero_f32_kernel(float* __restrict__ p, int n) {
  int gid = blockIdx.x * 256 + threadIdx.x;
  if (gid < n) p[gid] = 0.f;
}

// Layer-0 node transforms (K = 2): Ah/Bh/Dh/Eh = x @ W + b
__global__ void node_l0_kernel(const float* __restrict__ x,
                               const float* __restrict__ aw, const float* __restrict__ ab,
                               const float* __restrict__ bw, const float* __restrict__ bb,
                               const float* __restrict__ dw, const float* __restrict__ db,
                               const float* __restrict__ ew, const float* __restrict__ eb,
                               float* __restrict__ Ah, float* __restrict__ Bh,
                               float* __restrict__ Dh, float* __restrict__ Eh, int N) {
  int gid = blockIdx.x * 256 + threadIdx.x;
  if (gid >= N * HID) return;
  int n = gid >> 7, f = gid & 127;
  float x0 = x[n * 2 + 0], x1 = x[n * 2 + 1];
  Ah[gid] = x0 * aw[f] + x1 * aw[HID + f] + ab[f];
  Bh[gid] = x0 * bw[f] + x1 * bw[HID + f] + bb[f];
  Dh[gid] = x0 * dw[f] + x1 * dw[HID + f] + db[f];
  Eh[gid] = x0 * ew[f] + x1 * ew[HID + f] + eb[f];
}

// Edge update: e2 = Ce + Dh[src] + Eh[dst]; gate; atomic segment sums;
// e_out = leaky(act(bn(e2)) [+ e_in]). For layer 0, Ce is computed inline
// from the scalar edge feature (K = 1).
__global__ void edge_update_kernel(float* __restrict__ e2,               // in: Ce+Cb (l>=1), out: e'
                                   const float* __restrict__ efeat,      // layer0 raw [E,1]
                                   const float* __restrict__ c0w, const float* __restrict__ c0b,
                                   const float* __restrict__ Dh, const float* __restrict__ Eh,
                                   const float* __restrict__ Bh,
                                   const int* __restrict__ src, const int* __restrict__ dst,
                                   float* __restrict__ numb, float* __restrict__ denb,
                                   const float* __restrict__ e_in,
                                   const float* __restrict__ bng, const float* __restrict__ bnb,
                                   int E, int layer0) {
  int gid = blockIdx.x * 256 + threadIdx.x;
  if (gid >= E * HID) return;
  int k = gid >> 7, f = gid & 127;
  int s = src[k], d = dst[k];
  float v;
  if (layer0) v = efeat[k] * c0w[f] + c0b[f];
  else        v = e2[gid];
  v += Dh[s * HID + f] + Eh[d * HID + f];
  float sg = sigm(v);
  atomicAdd(&numb[d * HID + f], sg * Bh[s * HID + f]);
  atomicAdd(&denb[d * HID + f], sg);
  float t = bng[f] * (v * kBN_S) + bnb[f];
  if (layer0) { t = elu_f(t); }
  else        { t = lrelu(t); t += e_in[gid]; }
  e2[gid] = lrelu(t);
}

// Node finalize: h' = leaky(act(bn(Ah + num/(den+1e-6))) [+ h_in])
__global__ void node_finalize_kernel(const float* __restrict__ Ah,
                                     const float* __restrict__ numb, const float* __restrict__ denb,
                                     const float* __restrict__ h_in, float* __restrict__ h_out,
                                     const float* __restrict__ bng, const float* __restrict__ bnb,
                                     int N, int residual, int use_elu) {
  int gid = blockIdx.x * 256 + threadIdx.x;
  if (gid >= N * HID) return;
  int f = gid & 127;
  float v = Ah[gid] + numb[gid] / (denb[gid] + 1e-6f);
  v = bng[f] * (v * kBN_S) + bnb[f];
  v = use_elu ? elu_f(v) : lrelu(v);
  if (residual) v += h_in[gid];
  h_out[gid] = lrelu(v);
}

// Final classifier: out[E,2] = hid[E,128] @ w2[128,2] + b2
__global__ void mlp_out_kernel(const float* __restrict__ hid, const float* __restrict__ w2,
                               const float* __restrict__ b2, float* __restrict__ out, int E) {
  int k = blockIdx.x * 256 + threadIdx.x;
  if (k >= E) return;
  const float* hr = hid + (size_t)k * HID;
  float o0 = b2[0], o1 = b2[1];
#pragma unroll 8
  for (int j = 0; j < HID; ++j) {
    float v = hr[j];
    o0 += v * w2[j * 2 + 0];
    o1 += v * w2[j * 2 + 1];
  }
  out[k * 2 + 0] = o0;
  out[k * 2 + 1] = o1;
}

// ---------------------------------------------------------------------------
// Host orchestration
// ---------------------------------------------------------------------------
extern "C" void kernel_launch(void* const* d_in, const int* in_sizes, int n_in,
                              void* d_out, int out_size, void* d_ws, size_t ws_size,
                              hipStream_t stream) {
  const int N = NNODES, E = NEDGES;
  const float* node_feats = (const float*)d_in[0];
  const float* edge_feats = (const float*)d_in[1];
  const int*   src        = (const int*)d_in[2];
  const int*   dst        = (const int*)d_in[3];
  const float* a0_w = (const float*)d_in[4];   const float* a0_b = (const float*)d_in[5];
  const float* b0_w = (const float*)d_in[6];   const float* b0_b = (const float*)d_in[7];
  const float* c0_w = (const float*)d_in[8];   const float* c0_b = (const float*)d_in[9];
  const float* d0_w = (const float*)d_in[10];  const float* d0_b = (const float*)d_in[11];
  const float* e0_w = (const float*)d_in[12];  const float* e0_b = (const float*)d_in[13];
  const float* bnh0_g = (const float*)d_in[14]; const float* bnh0_b = (const float*)d_in[15];
  const float* bne0_g = (const float*)d_in[16]; const float* bne0_b = (const float*)d_in[17];
  const float* aL_w = (const float*)d_in[18];  const float* aL_b = (const float*)d_in[19];
  const float* bL_w = (const float*)d_in[20];  const float* bL_b = (const float*)d_in[21];
  const float* cL_w = (const float*)d_in[22];  const float* cL_b = (const float*)d_in[23];
  const float* dL_w = (const float*)d_in[24];  const float* dL_b = (const float*)d_in[25];
  const float* eL_w = (const float*)d_in[26];  const float* eL_b = (const float*)d_in[27];
  const float* bnhL_g = (const float*)d_in[28]; const float* bnhL_b = (const float*)d_in[29];
  const float* bneL_g = (const float*)d_in[30]; const float* bneL_b = (const float*)d_in[31];
  const float* mlp_w1 = (const float*)d_in[32]; const float* mlp_b1 = (const float*)d_in[33];
  const float* mlp_w2 = (const float*)d_in[34]; const float* mlp_b2 = (const float*)d_in[35];
  float* out = (float*)d_out;

  // --- workspace layout (floats) ---
  float* base = (float*)d_ws;
  size_t off = 0;
  auto fal = [&](size_t n) { float* p = base + off; off += n; return p; };
  const size_t NH = (size_t)N * HID, EH = (size_t)E * HID;
  float* hA  = fal(NH);
  float* hB  = fal(NH);
  float* Ah  = fal(NH);
  float* Bh  = fal(NH);
  float* Dh  = fal(NH);
  float* Eh  = fal(NH);
  float* num = fal(NH);
  float* den = fal(NH);   // contiguous after num (zeroed together)
  float* eA  = fal(EH);
  float* eB  = fal(EH);
  _Float16* wt = (_Float16*)(base + off);  // 512B-aligned (off multiple of 128)
  const size_t WSZ = (size_t)HID * HID;    // 16384 halfs per 128x128 weight
  _Float16* wtA[2] = { wt + 0 * WSZ, wt + 1 * WSZ };
  _Float16* wtB[2] = { wt + 2 * WSZ, wt + 3 * WSZ };
  _Float16* wtC[2] = { wt + 4 * WSZ, wt + 5 * WSZ };
  _Float16* wtD[2] = { wt + 6 * WSZ, wt + 7 * WSZ };
  _Float16* wtE[2] = { wt + 8 * WSZ, wt + 9 * WSZ };
  _Float16* wtM1   = wt + 10 * WSZ;        // [128][384]

  const int T = 256;
  const int gCvt128 = (HID * HID + T - 1) / T;        // 64
  const int gCvtMlp = (384 * HID + T - 1) / T;        // 192
  const int gNode   = (N * HID + T - 1) / T;          // 10000
  const int gEdge   = (E * HID + T - 1) / T;          // 160000
  const int gZero   = (2 * N * HID + T - 1) / T;      // 20000
  const int gGemmN  = ((N / 16) + 7) / 8;             // 157
  const int gGemmE  = ((E / 16) + 7) / 8;             // 2500
  const int gOut    = (E + T - 1) / T;                // 1250

  // 1) convert stacked weights to transposed f16
  for (int l = 0; l < 2; ++l) {
    cvt_wt_kernel<<<gCvt128, T, 0, stream>>>(aL_w + (size_t)l * WSZ, wtA[l], HID, HID);
    cvt_wt_kernel<<<gCvt128, T, 0, stream>>>(bL_w + (size_t)l * WSZ, wtB[l], HID, HID);
    cvt_wt_kernel<<<gCvt128, T, 0, stream>>>(cL_w + (size_t)l * WSZ, wtC[l], HID, HID);
    cvt_wt_kernel<<<gCvt128, T, 0, stream>>>(dL_w + (size_t)l * WSZ, wtD[l], HID, HID);
    cvt_wt_kernel<<<gCvt128, T, 0, stream>>>(eL_w + (size_t)l * WSZ, wtE[l], HID, HID);
  }
  cvt_wt_kernel<<<gCvtMlp, T, 0, stream>>>(mlp_w1, wtM1, 384, HID);

  // 2) layer 0 (2->128, 1->128): scalar transforms, ELU, no residual
  node_l0_kernel<<<gNode, T, 0, stream>>>(node_feats, a0_w, a0_b, b0_w, b0_b,
                                          d0_w, d0_b, e0_w, e0_b, Ah, Bh, Dh, Eh, N);
  zero_f32_kernel<<<gZero, T, 0, stream>>>(num, 2 * N * HID);
  edge_update_kernel<<<gEdge, T, 0, stream>>>(eA, edge_feats, c0_w, c0_b, Dh, Eh, Bh,
                                              src, dst, num, den, nullptr,
                                              bne0_g, bne0_b, E, /*layer0=*/1);
  node_finalize_kernel<<<gNode, T, 0, stream>>>(Ah, num, den, nullptr, hA,
                                                bnh0_g, bnh0_b, N, /*res=*/0, /*elu=*/1);

  // 3) stacked layers 1..2 (128->128, leaky inside, residual)
  float* h_cur = hA; float* h_nxt = hB;
  float* e_cur = eA; float* e_nxt = eB;
  for (int l = 0; l < 2; ++l) {
    const float* ab = aL_b + (size_t)l * HID;
    const float* bb = bL_b + (size_t)l * HID;
    const float* cb = cL_b + (size_t)l * HID;
    const float* db = dL_b + (size_t)l * HID;
    const float* eb = eL_b + (size_t)l * HID;
    gemm128_wmma<<<gGemmN, T, 0, stream>>>(h_cur, wtA[l], ab, Ah, N);
    gemm128_wmma<<<gGemmN, T, 0, stream>>>(h_cur, wtB[l], bb, Bh, N);
    gemm128_wmma<<<gGemmN, T, 0, stream>>>(h_cur, wtD[l], db, Dh, N);
    gemm128_wmma<<<gGemmN, T, 0, stream>>>(h_cur, wtE[l], eb, Eh, N);
    gemm128_wmma<<<gGemmE, T, 0, stream>>>(e_cur, wtC[l], cb, e_nxt, E);  // Ce + Cb
    zero_f32_kernel<<<gZero, T, 0, stream>>>(num, 2 * N * HID);
    edge_update_kernel<<<gEdge, T, 0, stream>>>(e_nxt, nullptr, nullptr, nullptr,
                                                Dh, Eh, Bh, src, dst, num, den, e_cur,
                                                bneL_g + (size_t)l * HID,
                                                bneL_b + (size_t)l * HID, E, /*layer0=*/0);
    node_finalize_kernel<<<gNode, T, 0, stream>>>(Ah, num, den, h_cur, h_nxt,
                                                  bnhL_g + (size_t)l * HID,
                                                  bnhL_b + (size_t)l * HID, N,
                                                  /*res=*/1, /*elu=*/0);
    float* t;
    t = h_cur; h_cur = h_nxt; h_nxt = t;
    t = e_cur; e_cur = e_nxt; e_nxt = t;
  }

  // 4) edge MLP: hid = relu(concat(h[src], h[dst], e) @ W1 + b1); out = hid @ W2 + b2
  float* hid = e_nxt;  // free ping-pong buffer
  gemm_mlp_wmma<<<gGemmE, T, 0, stream>>>(h_cur, e_cur, src, dst, wtM1, mlp_b1, hid, E);
  mlp_out_kernel<<<gOut, T, 0, stream>>>(hid, mlp_w2, mlp_b2, out, E);
}
